// SimplifiedSemanticModel_51539608360
// MI455X (gfx1250) — matrix operand order repaired and checked by reading, not compiled, not run on hardware
//
#include <hip/hip_runtime.h>

typedef __attribute__((ext_vector_type(16))) _Float16 v16h;
typedef __attribute__((ext_vector_type(8)))  float    v8f;

#define NPTS 12000
#define KNN  20

// Fragment-layout helpers (ISA 7.12.2, wave32):
// A (16x32 f16): lane m (0-15) holds K {0..7,16..23}; lane m+16 holds K {8..15,24..31}
// B (32x16 f16): lane n (0-15) holds K {0..7,16..23}; lane n+16 holds K {8..15,24..31}
__device__ __forceinline__ int a_lane(int m, int kl) { return m + (((kl >> 3) & 1) << 4); }
__device__ __forceinline__ int a_elem(int kl)        { return (kl & 7) + ((kl >= 16) ? 8 : 0); }
__device__ __forceinline__ int b_lane(int n, int kl) { return n + (((kl >> 3) & 1) << 4); }
__device__ __forceinline__ int b_elem(int kl)        { return (kl & 7) + ((kl >= 16) ? 8 : 0); }

// ---------------------------------------------------------------------------
__global__ void build_x0_kernel(const float* __restrict__ pts,
                                const float* __restrict__ feat,
                                float* __restrict__ x0, int n) {
  int i = blockIdx.x * blockDim.x + threadIdx.x;
  if (i < n) {
    x0[i*4+0] = pts[i*3+0];
    x0[i*4+1] = pts[i*3+1];
    x0[i*4+2] = pts[i*3+2];
    x0[i*4+3] = feat[i];
  }
}

// ---------------------------------------------------------------------------
template<int C, int LDA>
__global__ void sqnorm_kernel(const float* __restrict__ X,
                              float* __restrict__ sq, int n) {
  int i = blockIdx.x * blockDim.x + threadIdx.x;
  if (i < n) {
    float s = 0.f;
#pragma unroll
    for (int c = 0; c < C; ++c) { float v = X[(size_t)i*LDA + c]; s += v*v; }
    sq[i] = s;
  }
}

// ---------------------------------------------------------------------------
// wprep[0..63][c] = wA[o][c] ; wprep[64..127][c] = (wB - wA)[o][c]
// ---------------------------------------------------------------------------
template<int C>
__global__ void prep_w_kernel(const float* __restrict__ w,
                              float* __restrict__ wprep) {
  int t = blockIdx.x * blockDim.x + threadIdx.x;
  if (t < 64 * C) {
    int o = t / C, c = t % C;
    float a  = w[o*2*C + c];
    float bb = w[o*2*C + C + c];
    wprep[o*C + c]        = a;
    wprep[(64 + o)*C + c] = bb - a;
  }
}

// ---------------------------------------------------------------------------
// Fused kNN: block = 16 query rows, candidate tiles of 64 columns.
// dist = sq_q + sq_c - 2*(q.c) via v_wmma_f32_16x16x32_f16.
// All trip counts constexpr except the tile loop (unroll-disabled).
// ---------------------------------------------------------------------------
template<int C, int LDA>
__global__ __launch_bounds__(128) void knn_kernel(const float* __restrict__ X,
                                                  const float* __restrict__ sq,
                                                  int* __restrict__ knn_out, int n) {
  constexpr int KP  = (C + 31) & ~31;   // 32 or 64
  constexpr int NCH = KP / 32;          // 1 or 2

  __shared__ v16h AqF[NCH][32];         // [chunk][lane]
  __shared__ v16h BcF[NCH][4][32];      // [chunk][ntile][lane]
  __shared__ float Dist[16][64];
  __shared__ float SqQ[16];
  __shared__ float SqC[64];
  __shared__ float Ld[128 * KNN];
  __shared__ int   Li[128 * KNN];

  _Float16* Ap = (_Float16*)AqF;
  _Float16* Bp = (_Float16*)BcF;

  const int t    = threadIdx.x;
  const int lane = t & 31;
  const int wave = t >> 5;
  const int m0   = blockIdx.x * 16;

  // Stage queries in fragment layout (zero-padded).
#pragma unroll
  for (int e0 = 0; e0 < 16*KP; e0 += 128) {
    int e = e0 + t;
    int m = e / KP, k = e % KP;
    float v = 0.f;
    int gr = m0 + m;
    if (k < C && gr < n) v = X[(size_t)gr*LDA + k];
    int ch = k >> 5, kl = k & 31;
    Ap[(ch*32 + a_lane(m, kl))*16 + a_elem(kl)] = (_Float16)v;
  }
  if (t < 16) {
    int gr = m0 + t;
    SqQ[t] = (gr < n) ? sq[gr] : 0.f;
  }
#pragma unroll
  for (int k = 0; k < KNN; ++k) { Ld[t*KNN + k] = 3.0e38f; Li[t*KNN + k] = 0; }
  float curmax = 3.0e38f; int curpos = 0;
  __syncthreads();

  v16h afr[NCH];
#pragma unroll
  for (int ch = 0; ch < NCH; ++ch) afr[ch] = AqF[ch][lane];

#pragma clang loop unroll(disable)
  for (int j0 = 0; j0 < n; j0 += 64) {
    // Stage 64-candidate tile in B fragment layout.
#pragma unroll
    for (int e0 = 0; e0 < 64*KP; e0 += 128) {
      int e = e0 + t;
      int col = e / KP, k = e % KP;
      float v = 0.f;
      int gr = j0 + col;
      if (k < C && gr < n) v = X[(size_t)gr*LDA + k];
      int ch = k >> 5, kl = k & 31;
      int w = col >> 4, nn = col & 15;
      Bp[((ch*4 + w)*32 + b_lane(nn, kl))*16 + b_elem(kl)] = (_Float16)v;
    }
    if (t < 64) {
      int gr = j0 + t;
      SqC[t] = (gr < n) ? sq[gr] : 3.0e38f;
      int nr = gr + 64;
      if (nr < n) __builtin_prefetch(&X[(size_t)nr * LDA], 0, 3);
    }
    __syncthreads();

    // Each wave computes one 16x16 dot tile (wave w -> columns 16w..16w+15).
    v8f acc = {};
#pragma unroll
    for (int ch = 0; ch < NCH; ++ch) {
      v16h bfrag = BcF[ch][wave][lane];
      acc = __builtin_amdgcn_wmma_f32_16x16x32_f16(false, afr[ch], false, bfrag,
                                                   (short)0, acc, false, false);
    }
    {
      int mb = (lane >> 4) * 8;
      int nl = wave*16 + (lane & 15);
#pragma unroll
      for (int r = 0; r < 8; ++r) {
        int m = mb + r;
        Dist[m][nl] = SqQ[m] + SqC[nl] - 2.0f * acc[r];
      }
    }
    __syncthreads();

    // Running top-k: 8 threads per row strided over 64 columns.
    {
      int r = t >> 3, cs = t & 7;
#pragma clang loop unroll(disable)
      for (int cc = cs; cc < 64; cc += 8) {
        int j = j0 + cc;
        if (j >= n) continue;
        float d = Dist[r][cc];
        if (d < curmax) {
          Ld[t*KNN + curpos] = d;
          Li[t*KNN + curpos] = j;
          curmax = -3.0e38f;
#pragma clang loop unroll(disable)
          for (int k = 0; k < KNN; ++k) {
            float dk = Ld[t*KNN + k];
            if (dk > curmax) { curmax = dk; curpos = k; }
          }
        }
      }
    }
    __syncthreads();
  }

  // Merge 8 partial lists per row (contiguous 160 entries) -> 20 smallest.
  if (t < 16) {
    int gr = m0 + t;
    if (gr < n) {
      int base = t * 8 * KNN;
      for (int kk = 0; kk < KNN; ++kk) {
        float best = 3.0e38f; int bj = 0, bpos = -1;
#pragma clang loop unroll(disable)
        for (int q = 0; q < 8*KNN; ++q) {
          float d = Ld[base + q];
          if (d < best) { best = d; bj = Li[base + q]; bpos = q; }
        }
        if (bpos >= 0) Ld[base + bpos] = 3.1e38f;
        knn_out[gr*KNN + kk] = bj;
      }
    }
  }
}

// ---------------------------------------------------------------------------
// Generic WMMA GEMM: Out[M, 0..NOUT) = epi(A[M,K] @ B[NOUT,K]^T)
// Block tile 16 x 64 (4 waves x one 16x16 tile). K-loop fully unrolled
// (constexpr trip count) -> literal LDS offsets, no register rotation.
// EPI: 0 none | 1 leaky(v*g+b,0.2) | 2 v*g+b | 3 v+b
// ---------------------------------------------------------------------------
template<int K, int LDA, int LDB, int LDO, int NOUT, int EPI>
__global__ __launch_bounds__(128) void gemm_kernel(
    const float* __restrict__ A, const float* __restrict__ B,
    float* __restrict__ Out, int M,
    const float* __restrict__ g, const float* __restrict__ b) {
  constexpr int KP = (K + 31) & ~31;

  __shared__ v16h AsF[32];       // [lane]
  __shared__ v16h BsF[4][32];    // [ntile][lane]
  _Float16* Ap = (_Float16*)AsF;
  _Float16* Bp = (_Float16*)BsF;

  const int t    = threadIdx.x;
  const int lane = t & 31;
  const int wave = t >> 5;
  const int m0   = blockIdx.x * 16;
  const int n0   = blockIdx.y * 64;

  v8f acc = {};
#pragma unroll
  for (int kc = 0; kc < KP; kc += 32) {
    // Stage A chunk (16 x 32) in fragment layout.
#pragma unroll
    for (int e0 = 0; e0 < 16*32; e0 += 128) {
      int e = e0 + t;
      int m = e >> 5, kl = e & 31;
      float v = 0.f;
      int gr = m0 + m, gk = kc + kl;
      if (gk < K && gr < M) v = A[(size_t)gr*LDA + gk];
      Ap[a_lane(m, kl)*16 + a_elem(kl)] = (_Float16)v;
    }
    // Stage B chunk (64 x 32) in fragment layout.
#pragma unroll
    for (int e0 = 0; e0 < 64*32; e0 += 128) {
      int e = e0 + t;
      int col = e >> 5, kl = e & 31;
      float v = 0.f;
      int go2 = n0 + col, gk = kc + kl;
      if (gk < K && go2 < NOUT) v = B[(size_t)go2*LDB + gk];
      int w = col >> 4, nn = col & 15;
      Bp[(w*32 + b_lane(nn, kl))*16 + b_elem(kl)] = (_Float16)v;
    }
    // Prefetch next K chunk into L2.
    if (kc + 32 < KP) {
      if (t < 16 && m0 + t < M)    __builtin_prefetch(&A[(size_t)(m0 + t)*LDA + kc + 32], 0, 3);
      if (t < 64 && n0 + t < NOUT) __builtin_prefetch(&B[(size_t)(n0 + t)*LDB + kc + 32], 0, 3);
    }
    __syncthreads();

    v16h af = AsF[lane];
    v16h bf = BsF[wave][lane];
    acc = __builtin_amdgcn_wmma_f32_16x16x32_f16(false, af, false, bf,
                                                 (short)0, acc, false, false);
    __syncthreads();
  }

  // C layout: VGPR r -> M = r + 8*(lane>=16), N = lane&15
  int mb  = (lane >> 4) * 8;
  int col = n0 + wave*16 + (lane & 15);
#pragma unroll
  for (int r = 0; r < 8; ++r) {
    int row = m0 + mb + r;
    if (row < M && col < NOUT) {
      float v = acc[r];
      if (EPI == 1)      { v = v * g[col] + b[col]; v = (v >= 0.f) ? v : 0.2f*v; }
      else if (EPI == 2) { v = v * g[col] + b[col]; }
      else if (EPI == 3) { v = v + b[col]; }
      Out[(size_t)row*LDO + col] = v;
    }
  }
}

// ---------------------------------------------------------------------------
// Edge reduce: y[n,o] = leaky(((g>=0 ? max_k : min_k) u[idx[n,k],o] + t[n,o])*g + b)
// ---------------------------------------------------------------------------
__global__ __launch_bounds__(64) void edge_reduce_kernel(
    const float* __restrict__ ut, const int* __restrict__ knn,
    const float* __restrict__ g, const float* __restrict__ b,
    float* __restrict__ xcat, int colbase, int n) {
  __shared__ int nb[KNN];
  int nq = blockIdx.x;
  int o  = threadIdx.x;
  if (nq >= n) return;
  if (o < KNN) nb[o] = knn[nq*KNN + o];
  __syncthreads();

  float tt = ut[(size_t)nq*128 + 64 + o];
  float mx = -3.0e38f, mn = 3.0e38f;
#pragma unroll
  for (int k = 0; k < KNN; ++k) {
    float u = ut[(size_t)nb[k]*128 + o];
    mx = fmaxf(mx, u);
    mn = fminf(mn, u);
  }
  float gg = g[o], bb = b[o];
  float z = ((gg >= 0.f) ? mx : mn) + tt;
  float v = z * gg + bb;
  v = (v >= 0.f) ? v : 0.2f*v;
  xcat[(size_t)nq*256 + colbase + o] = v;
}

// ---------------------------------------------------------------------------
extern "C" void kernel_launch(void* const* d_in, const int* in_sizes, int n_in,
                              void* d_out, int out_size, void* d_ws, size_t ws_size,
                              hipStream_t stream) {
  const int N = NPTS;
  const float* points   = (const float*)d_in[0];
  const float* features = (const float*)d_in[1];
  const float* W[4]  = {(const float*)d_in[2], (const float*)d_in[5],
                        (const float*)d_in[8], (const float*)d_in[11]};
  const float* G[4]  = {(const float*)d_in[3], (const float*)d_in[6],
                        (const float*)d_in[9], (const float*)d_in[12]};
  const float* Bb[4] = {(const float*)d_in[4], (const float*)d_in[7],
                        (const float*)d_in[10], (const float*)d_in[13]};
  const float* w5 = (const float*)d_in[14];
  const float* g5 = (const float*)d_in[15];
  const float* b5 = (const float*)d_in[16];
  const float* wf = (const float*)d_in[17];
  const float* go = (const float*)d_in[18];
  const float* bo = (const float*)d_in[19];
  const float* wh = (const float*)d_in[20];
  const float* bh = (const float*)d_in[21];
  float* out = (float*)d_out;

  // Workspace carve-up (floats): ~81 MB total.
  float* ws    = (float*)d_ws;
  float* x0    = ws;                            // N*4
  float* xcat  = x0 + (size_t)N*4;              // N*256
  float* sq    = xcat + (size_t)N*256;          // N
  float* wprep = sq + N;                        // 128*64
  float* ut    = wprep + 128*64;                // N*128
  float* y5    = ut + (size_t)N*128;            // N*1024
  float* featb = y5 + (size_t)N*1024;           // N*256
  int*   knn   = (int*)(featb + (size_t)N*256); // N*20 ints

  build_x0_kernel<<<(N + 255)/256, 256, 0, stream>>>(points, features, x0, N);

  const int mblocks = (N + 15)/16;   // 750

  for (int i = 0; i < 4; ++i) {
    if (i == 0) {
      sqnorm_kernel<4,4><<<(N + 255)/256, 256, 0, stream>>>(x0, sq, N);
      knn_kernel<4,4><<<mblocks, 128, 0, stream>>>(x0, sq, knn, N);
      prep_w_kernel<4><<<1, 256, 0, stream>>>(W[0], wprep);
      gemm_kernel<4,4,4,128,128,0><<<dim3(mblocks, 2), 128, 0, stream>>>(
          x0, wprep, ut, N, nullptr, nullptr);
    } else {
      const float* Xi = xcat + (size_t)(i - 1)*64;
      sqnorm_kernel<64,256><<<(N + 255)/256, 256, 0, stream>>>(Xi, sq, N);
      knn_kernel<64,256><<<mblocks, 128, 0, stream>>>(Xi, sq, knn, N);
      prep_w_kernel<64><<<16, 256, 0, stream>>>(W[i], wprep);
      gemm_kernel<64,256,64,128,128,0><<<dim3(mblocks, 2), 128, 0, stream>>>(
          Xi, wprep, ut, N, nullptr, nullptr);
    }
    edge_reduce_kernel<<<N, 64, 0, stream>>>(ut, knn, G[i], Bb[i], xcat, i*64, N);
  }

  // conv5: y5 = leaky(xcat @ w5^T * g5 + b5)
  gemm_kernel<256,256,256,1024,1024,1><<<dim3(mblocks, 16), 128, 0, stream>>>(
      xcat, w5, y5, N, g5, b5);
  // feat = (y5 @ wf^T) * go + bo
  gemm_kernel<1024,1024,1024,256,256,2><<<dim3(mblocks, 4), 128, 0, stream>>>(
      y5, wf, featb, N, go, bo);
  // head = feat @ wh^T + bh
  gemm_kernel<256,256,256,20,20,3><<<dim3(mblocks, 1), 128, 0, stream>>>(
      featb, wh, out, N, nullptr, bh);
}
